// SelectiveScanIC_71777493451116
// MI455X (gfx1250) — matrix operand order repaired and checked by reading, not compiled, not run on hardware
//
#include <hip/hip_runtime.h>

typedef __attribute__((ext_vector_type(16))) __bf16          v16bf;
typedef __attribute__((ext_vector_type(8)))  float           v8f;
typedef __attribute__((ext_vector_type(4)))  float           v4f;
typedef __attribute__((ext_vector_type(8)))  unsigned short  v8u16;
typedef __attribute__((ext_vector_type(16))) unsigned short  v16u16;

#define B_DIM 4
#define K_DIM 8192
#define D_DIM 1024
#define M_TOT (B_DIM * K_DIM)   // 32768 rows of x flattened [B*K, D]

// ---------------- GEMM tile config ----------------
#define TM  128      // rows per block
#define TN  64       // cols per block
#define TK  64       // K stage depth
#define TKP 88       // padded LDS row (88*2B = 176B rows: 16B-aligned, bank-spread)

__device__ __forceinline__ unsigned short f2bf(float f) {
    union { float f; unsigned int u; } v; v.f = f;
    unsigned int u = v.u;
    // round-to-nearest-even bf16
    return (unsigned short)((u + 0x7FFFu + ((u >> 16) & 1u)) >> 16);
}

__device__ __forceinline__ v16bf frag_from(const unsigned short* p0,
                                           const unsigned short* p1) {
    v8u16 lo = *(const v8u16*)p0;
    v8u16 hi = *(const v8u16*)p1;
    v16u16 q = __builtin_shufflevector(lo, hi, 0,1,2,3,4,5,6,7,
                                               8,9,10,11,12,13,14,15);
    return __builtin_bit_cast(v16bf, q);
}

// lam = sigmoid(x @ W^T + b), written f32 to `lam` ( == d_out, overwritten later)
__global__ __launch_bounds__(256)
void gemm_sigmoid_kernel(const float* __restrict__ x,
                         const float* __restrict__ W,
                         const float* __restrict__ bias,
                         float* __restrict__ lam) {
    __shared__ __align__(16) unsigned short As[TM][TKP]; // 128 x (64 used)
    __shared__ __align__(16) unsigned short Bs[TN][TKP]; // 64  x (64 used)

    const int tid   = threadIdx.x;
    const int lane  = tid & 31;
    const int wave  = tid >> 5;          // 0..7
    const int wm    = wave & 3;          // 4 waves along M  -> 32 rows each
    const int wn    = wave >> 2;         // 2 waves along N  -> 32 cols each
    const int lrow  = lane & 15;
    const int lhi   = lane >> 4;         // 0/1: which half of the wave

    const int m0 = blockIdx.y * TM;      // global row base
    const int n0 = blockIdx.x * TN;      // global col base

    v8f acc[2][2];
    #pragma unroll
    for (int i = 0; i < 2; ++i)
        #pragma unroll
        for (int j = 0; j < 2; ++j)
            acc[i][j] = (v8f)0.0f;

    for (int k0 = 0; k0 < D_DIM; k0 += TK) {
        // ---- stage A: 128x64 f32 -> bf16 in LDS (2x b128 load -> 1x b128 store) ----
        #pragma unroll
        for (int j = 0; j < 4; ++j) {
            int vec  = j * 256 + tid;          // 0..1023 (8-elem groups)
            int row  = vec >> 3;               // 0..127
            int col8 = (vec & 7) << 3;         // 0,8,..,56
            const float* src = &x[(size_t)(m0 + row) * D_DIM + k0 + col8];
            v4f a0 = *(const v4f*)src;
            v4f a1 = *(const v4f*)(src + 4);
            v8u16 p = { f2bf(a0.x), f2bf(a0.y), f2bf(a0.z), f2bf(a0.w),
                        f2bf(a1.x), f2bf(a1.y), f2bf(a1.z), f2bf(a1.w) };
            *(v8u16*)&As[row][col8] = p;
        }
        // ---- stage B: Bs[n][k] = W[n0+n][k0+k] (B = W^T, rows of W contiguous) ----
        #pragma unroll
        for (int j = 0; j < 2; ++j) {
            int vec  = j * 256 + tid;          // 0..511
            int row  = vec >> 3;               // 0..63
            int col8 = (vec & 7) << 3;
            const float* src = &W[(size_t)(n0 + row) * D_DIM + k0 + col8];
            v4f b0 = *(const v4f*)src;
            v4f b1 = *(const v4f*)(src + 4);
            v8u16 p = { f2bf(b0.x), f2bf(b0.y), f2bf(b0.z), f2bf(b0.w),
                        f2bf(b1.x), f2bf(b1.y), f2bf(b1.z), f2bf(b1.w) };
            *(v8u16*)&Bs[row][col8] = p;
        }
        __syncthreads();

        // ---- 2 k-substeps of 32, 2x2 fragments per wave ----
        #pragma unroll
        for (int ks = 0; ks < 2; ++ks) {
            const int kb = ks * 32;
            v16bf afr[2], bfr[2];
            #pragma unroll
            for (int mf = 0; mf < 2; ++mf) {
                int ar = wm * 32 + mf * 16 + lrow;
                // A 16x32 bf16 layout: lane half lhi holds K {lhi*8..+7, 16+lhi*8..+7}
                afr[mf] = frag_from(&As[ar][kb + lhi * 8],
                                    &As[ar][kb + 16 + lhi * 8]);
            }
            #pragma unroll
            for (int nf = 0; nf < 2; ++nf) {
                int bc = wn * 32 + nf * 16 + lrow;
                // B 32x16 bf16 layout: lane half lhi holds K {lhi*8..+7, 16+lhi*8..+7}
                bfr[nf] = frag_from(&Bs[bc][kb + lhi * 8],
                                    &Bs[bc][kb + 16 + lhi * 8]);
            }
            #pragma unroll
            for (int mf = 0; mf < 2; ++mf)
                #pragma unroll
                for (int nf = 0; nf < 2; ++nf)
                    acc[mf][nf] = __builtin_amdgcn_wmma_f32_16x16x32_bf16(
                        false, afr[mf], false, bfr[nf],
                        (short)0, acc[mf][nf], false, false);
        }
        __syncthreads();
    }

    // ---- epilogue: + bias, sigmoid, store f32 ----
    #pragma unroll
    for (int nf = 0; nf < 2; ++nf) {
        const int col = n0 + wn * 32 + nf * 16 + lrow;
        const float bv = bias[col];
        #pragma unroll
        for (int mf = 0; mf < 2; ++mf) {
            #pragma unroll
            for (int r = 0; r < 8; ++r) {
                int m = m0 + wm * 32 + mf * 16 + r + 8 * lhi;
                float z = acc[mf][nf][r] + bv;
                float l = 1.0f / (1.0f + __expf(-z));
                lam[(size_t)m * D_DIM + col] = l;
            }
        }
    }
}

// ---------------- chunked selective scan ----------------
// block = (batch b, group of 32 channels); 8 waves = 8 k-chunks of 1024 steps.
// io holds lam on entry (from GEMM), and the scan result on exit (in place).
#define KCH 8
#define CHG 32
__global__ __launch_bounds__(256)
void scan_kernel(const float* __restrict__ x, float* __restrict__ io) {
    const int blk  = blockIdx.x;              // 0 .. B*(D/32)-1
    const int b    = blk / (D_DIM / CHG);
    const int dg   = blk % (D_DIM / CHG);
    const int t    = threadIdx.x;
    const int dloc = t & 31;
    const int ch   = dg * CHG + dloc;         // channel d
    const int kc   = t >> 5;                  // chunk id 0..7
    const int CL   = K_DIM / KCH;             // 1024 steps per chunk

    const size_t base = ((size_t)b * K_DIM) * D_DIM + ch;
    const size_t idx0 = base + (size_t)(kc * CL) * D_DIM;

    // phase 1: per-chunk affine composition  s_out = A*s_in + C
    float A = 1.0f, C = 0.0f;
    size_t idx = idx0;
    for (int i = 0; i < CL; ++i) {
        float l  = io[idx];
        float xv = x[idx];
        A = l * A;
        C = fmaf(l, C, (1.0f - l) * xv);
        idx += D_DIM;
    }

    __shared__ float sA[KCH][CHG];
    __shared__ float sC[KCH][CHG];
    sA[kc][dloc] = A;
    sC[kc][dloc] = C;
    __syncthreads();

    // phase 2: exclusive scan over the 8 chunks (wave 0, one lane per channel)
    if (t < CHG) {
        float a = 1.0f, c = 0.0f;             // identity (s0 = 0)
        #pragma unroll
        for (int j = 0; j < KCH; ++j) {
            float aj = sA[j][t], cj = sC[j][t];
            sA[j][t] = a; sC[j][t] = c;       // exclusive prefix
            float na = aj * a;
            float nc = fmaf(aj, c, cj);
            a = na; c = nc;
        }
    }
    __syncthreads();

    // phase 3: replay with incoming state (applied to s0=0 the prefix gives C)
    float s = sC[kc][dloc];
    idx = idx0;
    for (int i = 0; i < CL; ++i) {
        float l  = io[idx];
        float xv = x[idx];
        s = fmaf(l, s, (1.0f - l) * xv);
        io[idx] = s;
        idx += D_DIM;
    }
}

extern "C" void kernel_launch(void* const* d_in, const int* in_sizes, int n_in,
                              void* d_out, int out_size, void* d_ws, size_t ws_size,
                              hipStream_t stream) {
    const float* x    = (const float*)d_in[0];   // [B,K,D]
    const float* W    = (const float*)d_in[1];   // [D,D]
    const float* bias = (const float*)d_in[2];   // [D]
    float* out = (float*)d_out;                  // [B,K,D]
    (void)in_sizes; (void)n_in; (void)out_size; (void)d_ws; (void)ws_size;

    dim3 g1(D_DIM / TN, M_TOT / TM);             // 16 x 256 blocks
    gemm_sigmoid_kernel<<<g1, dim3(256), 0, stream>>>(x, W, bias, out);

    dim3 g2(B_DIM * (D_DIM / CHG));              // 128 blocks
    scan_kernel<<<g2, dim3(256), 0, stream>>>(x, out);
}
